// GAT_14559939133705
// MI455X (gfx1250) — matrix operand order repaired and checked by reading, not compiled, not run on hardware
//
#include <hip/hip_runtime.h>
#include <hip/hip_bf16.h>

// ---------------------------------------------------------------------------
// Types for CDNA5 WMMA
// ---------------------------------------------------------------------------
typedef __bf16  v16bf __attribute__((ext_vector_type(16)));
typedef float   v8f   __attribute__((ext_vector_type(8)));

union FragV {
    unsigned int u[8];
    v16bf        v;
};

// ---------------------------------------------------------------------------
// Helpers
// ---------------------------------------------------------------------------
__device__ inline float leaky_relu(float x) { return x > 0.f ? x : 0.2f * x; }

// order-preserving float -> uint encoding for atomic max over signed floats
__device__ inline unsigned enc_f32(float v) {
    unsigned u = __float_as_uint(v);
    return (u & 0x80000000u) ? ~u : (u | 0x80000000u);
}
__device__ inline float dec_f32(unsigned e) {
    return (e & 0x80000000u) ? __uint_as_float(e & 0x7FFFFFFFu)
                             : __uint_as_float(~e);
}
// enc_f32(-inf) == ~0xFF800000 == 0x007FFFFF
#define ENC_NEG_INF 0x007FFFFFu

// gfx1250 async global->LDS copy (ASYNCcnt-tracked DMA), 16B per lane
__device__ inline void async_copy_b128(unsigned lds_off, const void* gaddr) {
    asm volatile("global_load_async_to_lds_b128 %0, %1, off"
                 :: "v"(lds_off), "v"(gaddr) : "memory");
}

// ---------------------------------------------------------------------------
// Elementwise utility kernels
// ---------------------------------------------------------------------------
__global__ void f2bf_kernel(const float* __restrict__ in, __bf16* __restrict__ out, long n) {
    long t = (long)blockIdx.x * blockDim.x + threadIdx.x;
    if (t < n) out[t] = (__bf16)in[t];
}

__global__ void fill_u32_kernel(unsigned* __restrict__ p, unsigned v, long n) {
    long t = (long)blockIdx.x * blockDim.x + threadIdx.x;
    if (t < n) p[t] = v;
}

__global__ void add_bias_kernel(float* __restrict__ a, const float* __restrict__ b,
                                long rows, int cols) {
    long t = (long)blockIdx.x * blockDim.x + threadIdx.x;
    if (t < rows * (long)cols) a[t] += b[t % cols];
}

// ---------------------------------------------------------------------------
// Tiled bf16 WMMA GEMM:  C[M,N] = A[M,K] * B[N,K]^T (+ bias[N])
// Block = 256 threads = 8 waves; tile 128x128; K-step 32.
// Interior blocks: double-buffered async global->LDS DMA overlapped with WMMA.
// Partial M blocks: synchronous staging with zero-fill.
// ---------------------------------------------------------------------------
#define TILE_M 128
#define TILE_N 128
#define LDT    40                  // padded LDS row stride (80 B: 16B aligned)
#define ABUF   (TILE_M * LDT)      // A-tile elems per buffer
#define BUFE   ((TILE_M + TILE_N) * LDT)   // elems per buffer (A then B)

// one 16x16x32 WMMA sweep over a staged 128x128x32 tile (per-wave 2x4 tiles)
__device__ inline void mma_step(const __bf16* bufA, const __bf16* bufB,
                                int wm, int wn, int hlf, int l16,
                                v8f acc[2][4]) {
    FragV fb[4];
    for (int tn = 0; tn < 4; ++tn) {
        int nrow = wn * 64 + tn * 16 + l16;
        for (int v = 0; v < 8; ++v) {
            int kk = hlf * 16 + v * 2;
            fb[tn].u[v] = *(const unsigned int*)(bufB + nrow * LDT + kk);
        }
    }
    for (int tm = 0; tm < 2; ++tm) {
        FragV fa;
        int mrow = wm * 32 + tm * 16 + l16;
        for (int v = 0; v < 8; ++v) {
            int kk = hlf * 16 + v * 2;
            fa.u[v] = *(const unsigned int*)(bufA + mrow * LDT + kk);
        }
        for (int tn = 0; tn < 4; ++tn) {
            acc[tm][tn] = __builtin_amdgcn_wmma_f32_16x16x32_bf16(
                false, fa.v, false, fb[tn].v,
                (short)0, acc[tm][tn], false, false);
        }
    }
}

__global__ __launch_bounds__(256)
void gemm_bf16_wmma(const __bf16* __restrict__ A,
                    const __bf16* __restrict__ B,
                    const float* __restrict__ bias,
                    float* __restrict__ C,
                    int M, int N, int K) {
    __shared__ __align__(16) __bf16 lds[2 * BUFE];

    const int tid  = threadIdx.x;
    const int lane = tid & 31;
    const int wv   = tid >> 5;     // 0..7
    const int wm   = wv & 3;       // 32-row strip within tile
    const int wn   = wv >> 2;      // 64-col strip within tile
    const int hlf  = lane >> 4;    // 0/1 (lane half)
    const int l16  = lane & 15;

    const int mBlock = blockIdx.x * TILE_M;
    const int nBlock = blockIdx.y * TILE_N;

    // staging geometry: thread -> (row, 8-elem segment); covers rows r and r+64
    const int rowa = tid >> 2;
    const int seg  = (tid & 3) * 8;

    v8f acc[2][4];
    for (int a = 0; a < 2; ++a)
        for (int b = 0; b < 4; ++b)
            for (int i = 0; i < 8; ++i) acc[a][b][i] = 0.f;

    const int steps = K / 32;
    const bool interior = (mBlock + TILE_M <= M);

    if (interior) {
        // -------- async double-buffered pipeline --------
        auto issue_tile = [&](int buf, int kb) {
            const __bf16* gA = A + (size_t)(mBlock + rowa) * K + kb + seg;
            const __bf16* gB = B + (size_t)(nBlock + rowa) * K + kb + seg;
            __bf16* lA = lds + buf * BUFE + rowa * LDT + seg;
            __bf16* lB = lds + buf * BUFE + ABUF + rowa * LDT + seg;
            async_copy_b128((unsigned)(uintptr_t)lA, (const void*)gA);
            async_copy_b128((unsigned)(uintptr_t)(lA + 64 * LDT),
                            (const void*)(gA + (size_t)64 * K));
            async_copy_b128((unsigned)(uintptr_t)lB, (const void*)gB);
            async_copy_b128((unsigned)(uintptr_t)(lB + 64 * LDT),
                            (const void*)(gB + (size_t)64 * K));
        };

        issue_tile(0, 0);
        if (steps > 1) issue_tile(1, 32);

        for (int s = 0; s < steps; ++s) {
            // wait for tile s to land (tile s+1 may still be in flight: 4 ops)
            if (s + 1 < steps)
                asm volatile("s_wait_asynccnt 0x4" ::: "memory");
            else
                asm volatile("s_wait_asynccnt 0x0" ::: "memory");
            __syncthreads();
            const __bf16* bufA = lds + (s & 1) * BUFE;
            mma_step(bufA, bufA + ABUF, wm, wn, hlf, l16, acc);
            __syncthreads();
            if (s + 2 < steps) issue_tile(s & 1, (s + 2) * 32);
        }
    } else {
        // -------- synchronous fallback for partial M blocks --------
        for (int kb = 0; kb < K; kb += 32) {
            for (int i = 0; i < 2; ++i) {
                int row = rowa + i * 64;
                uint4 val = make_uint4(0u, 0u, 0u, 0u);
                int gr = mBlock + row;
                if (gr < M)
                    val = *(const uint4*)(A + (size_t)gr * K + kb + seg);
                *(uint4*)(lds + row * LDT + seg) = val;
            }
            for (int i = 0; i < 2; ++i) {
                int row = rowa + i * 64;
                uint4 val = *(const uint4*)(B + (size_t)(nBlock + row) * K + kb + seg);
                *(uint4*)(lds + ABUF + row * LDT + seg) = val;
            }
            __syncthreads();
            mma_step(lds, lds + ABUF, wm, wn, hlf, l16, acc);
            __syncthreads();
        }
    }

    // Epilogue: C/D layout VGPR r -> M = r + (laneHalf)*8 ; col = lane%16
    for (int tm = 0; tm < 2; ++tm) {
        for (int tn = 0; tn < 4; ++tn) {
            int col = nBlock + wn * 64 + tn * 16 + l16;
            float bv = bias ? bias[col] : 0.f;
            for (int r = 0; r < 8; ++r) {
                int row = mBlock + wm * 32 + tm * 16 + hlf * 8 + r;
                if (row < M) C[(size_t)row * N + col] = acc[tm][tn][r] + bv;
            }
        }
    }
}

// ---------------------------------------------------------------------------
// Per-(node,head) attention score dots: es = h.a_src, ed = h.a_dst (wave-per-pair)
// ---------------------------------------------------------------------------
__global__ void node_scores_kernel(const float* __restrict__ h,
                                   const float* __restrict__ as,
                                   const float* __restrict__ ad,
                                   float* __restrict__ es, float* __restrict__ ed,
                                   int n, int H, int HC) {
    long gt = (long)blockIdx.x * blockDim.x + threadIdx.x;
    long w  = gt >> 5;
    int lane = threadIdx.x & 31;
    if (w >= (long)n * H) return;
    int node = (int)(w / H), head = (int)(w % H);
    const float* hp = h + (size_t)node * H * HC + (size_t)head * HC;
    float s = 0.f, d = 0.f;
    for (int c = lane; c < HC; c += 32) {
        float v = hp[c];
        s += v * as[head * HC + c];
        d += v * ad[head * HC + c];
    }
    for (int o = 16; o; o >>= 1) {
        s += __shfl_xor(s, o, 32);
        d += __shfl_xor(d, o, 32);
    }
    if (lane == 0) {
        es[(size_t)node * H + head] = s;
        ed[(size_t)node * H + head] = d;
    }
}

// ---------------------------------------------------------------------------
// Edge phase kernels (self-loops appended implicitly: edge >= E -> node loop)
// ---------------------------------------------------------------------------
__device__ inline void edge_src_dst(const int* ei, int E, int edge, int& s, int& d) {
    if (edge < E) { s = ei[edge]; d = ei[E + edge]; }
    else          { s = d = edge - E; }
}

__global__ void edge_max_kernel(const int* __restrict__ ei, int E, int Etot, int H,
                                const float* __restrict__ es, const float* __restrict__ ed,
                                unsigned* __restrict__ mx) {
    long t = (long)blockIdx.x * blockDim.x + threadIdx.x;
    if (t >= (long)Etot * H) return;
    int edge = (int)(t / H), head = (int)(t % H);
    int s, d; edge_src_dst(ei, E, edge, s, d);
    float v = leaky_relu(es[(size_t)s * H + head] + ed[(size_t)d * H + head]);
    atomicMax(&mx[(size_t)d * H + head], enc_f32(v));
}

__global__ void edge_expsum_kernel(const int* __restrict__ ei, int E, int Etot, int H,
                                   const float* __restrict__ es, const float* __restrict__ ed,
                                   const unsigned* __restrict__ mx, float* __restrict__ dn) {
    long t = (long)blockIdx.x * blockDim.x + threadIdx.x;
    if (t >= (long)Etot * H) return;
    int edge = (int)(t / H), head = (int)(t % H);
    int s, d; edge_src_dst(ei, E, edge, s, d);
    size_t di = (size_t)d * H + head;
    float v = leaky_relu(es[(size_t)s * H + head] + ed[di]);
    unsafeAtomicAdd(&dn[di], expf(v - dec_f32(mx[di])));
}

// wave per (edge, head); lanes stride the head channels
__global__ void edge_agg_kernel(const int* __restrict__ ei, int E, int Etot, int H, int HC,
                                const float* __restrict__ es, const float* __restrict__ ed,
                                const unsigned* __restrict__ mx, const float* __restrict__ dn,
                                const float* __restrict__ h, float* __restrict__ agg) {
    long gt = (long)blockIdx.x * blockDim.x + threadIdx.x;
    long w  = gt >> 5;
    int lane = threadIdx.x & 31;
    if (w >= (long)Etot * H) return;
    int edge = (int)(w / H), head = (int)(w % H);
    int s, d; edge_src_dst(ei, E, edge, s, d);
    size_t si = (size_t)s * H + head, di = (size_t)d * H + head;
    float v = leaky_relu(es[si] + ed[di]);
    float alpha = expf(v - dec_f32(mx[di])) / dn[di];
    const float* hp = h   + (size_t)s * H * HC + (size_t)head * HC;
    float*       op = agg + (size_t)d * H * HC + (size_t)head * HC;
    for (int c = lane; c < HC; c += 32)
        unsafeAtomicAdd(&op[c], alpha * hp[c]);
}

// ---------------------------------------------------------------------------
// Final: out = log_softmax(agg2 + b_last + skip), wave per node, C == 256
// ---------------------------------------------------------------------------
__global__ void final_logsoftmax_kernel(const float* __restrict__ agg2,
                                        const float* __restrict__ b,
                                        const float* __restrict__ skip,
                                        float* __restrict__ out, int n, int C) {
    long gt = (long)blockIdx.x * blockDim.x + threadIdx.x;
    int node = (int)(gt >> 5);
    int lane = threadIdx.x & 31;
    if (node >= n) return;
    float vals[8];                 // C/32 == 8
    int nc = C / 32;
    float mxv = -3.402823466e38f;
    for (int i = 0; i < nc; ++i) {
        int c = lane + i * 32;
        float v = agg2[(size_t)node * C + c] + b[c] + skip[(size_t)node * C + c];
        vals[i] = v;
        mxv = fmaxf(mxv, v);
    }
    for (int o = 16; o; o >>= 1) mxv = fmaxf(mxv, __shfl_xor(mxv, o, 32));
    float se = 0.f;
    for (int i = 0; i < nc; ++i) se += expf(vals[i] - mxv);
    for (int o = 16; o; o >>= 1) se += __shfl_xor(se, o, 32);
    float lse = logf(se) + mxv;
    for (int i = 0; i < nc; ++i)
        out[(size_t)node * C + lane + i * 32] = vals[i] - lse;
}

// ---------------------------------------------------------------------------
// Host-side orchestration
// ---------------------------------------------------------------------------
static inline char* carve(char*& p, size_t bytes) {
    char* r = p;
    p += (bytes + 255) & ~(size_t)255;
    return r;
}
static inline int cdiv(long a, long b) { return (int)((a + b - 1) / b); }

extern "C" void kernel_launch(void* const* d_in, const int* in_sizes, int n_in,
                              void* d_out, int out_size, void* d_ws, size_t ws_size,
                              hipStream_t stream) {
    (void)n_in; (void)out_size; (void)ws_size;
    const float* x        = (const float*)d_in[0];
    const int*   ei       = (const int*)  d_in[1];
    const float* W_heads  = (const float*)d_in[2];
    const float* a_src_h  = (const float*)d_in[3];
    const float* a_dst_h  = (const float*)d_in[4];
    const float* b_heads  = (const float*)d_in[5];
    const float* W_last   = (const float*)d_in[6];
    const float* a_src_l  = (const float*)d_in[7];
    const float* a_dst_l  = (const float*)d_in[8];
    const float* b_last   = (const float*)d_in[9];
    const float* W_skip   = (const float*)d_in[10];
    const float* b_skip   = (const float*)d_in[11];
    float* out = (float*)d_out;

    const int IN = 512, HID = 512, OUT = 256, H = 8, HC = 64;
    const int N = in_sizes[0] / IN;
    const int E = in_sizes[1] / 2;
    const int Etot = E + N;

    char* p = (char*)d_ws;
    __bf16*   x_bf  = (__bf16*)  carve(p, (size_t)N * IN * 2);
    __bf16*   Wh_bf = (__bf16*)  carve(p, (size_t)HID * IN * 2);
    __bf16*   Wl_bf = (__bf16*)  carve(p, (size_t)OUT * HID * 2);
    __bf16*   Ws_bf = (__bf16*)  carve(p, (size_t)OUT * IN * 2);
    float*    h1    = (float*)   carve(p, (size_t)N * HID * 4);
    float*    skip  = (float*)   carve(p, (size_t)N * OUT * 4);
    float*    es1   = (float*)   carve(p, (size_t)N * H * 4);
    float*    ed1   = (float*)   carve(p, (size_t)N * H * 4);
    unsigned* mx1   = (unsigned*)carve(p, (size_t)N * H * 4);
    float*    dn1   = (float*)   carve(p, (size_t)N * H * 4);
    float*    agg1  = (float*)   carve(p, (size_t)N * HID * 4);   // becomes h2
    __bf16*   h2_bf = (__bf16*)  carve(p, (size_t)N * HID * 2);
    float*    g2    = (float*)   carve(p, (size_t)N * OUT * 4);
    float*    es2   = (float*)   carve(p, (size_t)N * 4);
    float*    ed2   = (float*)   carve(p, (size_t)N * 4);
    unsigned* mx2   = (unsigned*)carve(p, (size_t)N * 4);
    float*    dn2   = (float*)   carve(p, (size_t)N * 4);
    float*    agg2  = (float*)   carve(p, (size_t)N * OUT * 4);

    // 1) fp32 -> bf16 operand conversion
    f2bf_kernel<<<cdiv((long)N * IN, 256), 256, 0, stream>>>(x, x_bf, (long)N * IN);
    f2bf_kernel<<<cdiv((long)HID * IN, 256), 256, 0, stream>>>(W_heads, Wh_bf, (long)HID * IN);
    f2bf_kernel<<<cdiv((long)OUT * HID, 256), 256, 0, stream>>>(W_last, Wl_bf, (long)OUT * HID);
    f2bf_kernel<<<cdiv((long)OUT * IN, 256), 256, 0, stream>>>(W_skip, Ws_bf, (long)OUT * IN);

    // 2) GEMMs: h1 = x Wh^T ; skip = x Ws^T + b_skip
    dim3 gHID(cdiv(N, TILE_M), HID / TILE_N);
    dim3 gOUT(cdiv(N, TILE_M), OUT / TILE_N);
    gemm_bf16_wmma<<<gHID, 256, 0, stream>>>(x_bf, Wh_bf, nullptr, h1, N, HID, IN);
    gemm_bf16_wmma<<<gOUT, 256, 0, stream>>>(x_bf, Ws_bf, b_skip, skip, N, OUT, IN);

    // 3) layer-1 attention
    node_scores_kernel<<<cdiv((long)N * H * 32, 256), 256, 0, stream>>>(
        h1, a_src_h, a_dst_h, es1, ed1, N, H, HC);
    fill_u32_kernel<<<cdiv((long)N * HID, 256), 256, 0, stream>>>((unsigned*)agg1, 0u, (long)N * HID);
    fill_u32_kernel<<<cdiv((long)N * H, 256), 256, 0, stream>>>(mx1, ENC_NEG_INF, (long)N * H);
    fill_u32_kernel<<<cdiv((long)N * H, 256), 256, 0, stream>>>((unsigned*)dn1, 0u, (long)N * H);
    edge_max_kernel<<<cdiv((long)Etot * H, 256), 256, 0, stream>>>(ei, E, Etot, H, es1, ed1, mx1);
    edge_expsum_kernel<<<cdiv((long)Etot * H, 256), 256, 0, stream>>>(ei, E, Etot, H, es1, ed1, mx1, dn1);
    edge_agg_kernel<<<cdiv((long)Etot * H * 32, 256), 256, 0, stream>>>(
        ei, E, Etot, H, HC, es1, ed1, mx1, dn1, h1, agg1);

    // 4) h2 = agg1 + b_heads ; convert to bf16 ; g2 = h2 Wl^T
    add_bias_kernel<<<cdiv((long)N * HID, 256), 256, 0, stream>>>(agg1, b_heads, N, HID);
    f2bf_kernel<<<cdiv((long)N * HID, 256), 256, 0, stream>>>(agg1, h2_bf, (long)N * HID);
    gemm_bf16_wmma<<<gOUT, 256, 0, stream>>>(h2_bf, Wl_bf, nullptr, g2, N, OUT, HID);

    // 5) layer-2 attention (single head, HC = OUT)
    node_scores_kernel<<<cdiv((long)N * 32, 256), 256, 0, stream>>>(
        g2, a_src_l, a_dst_l, es2, ed2, N, 1, OUT);
    fill_u32_kernel<<<cdiv((long)N * OUT, 256), 256, 0, stream>>>((unsigned*)agg2, 0u, (long)N * OUT);
    fill_u32_kernel<<<cdiv((long)N, 256), 256, 0, stream>>>(mx2, ENC_NEG_INF, (long)N);
    fill_u32_kernel<<<cdiv((long)N, 256), 256, 0, stream>>>((unsigned*)dn2, 0u, (long)N);
    edge_max_kernel<<<cdiv((long)Etot, 256), 256, 0, stream>>>(ei, E, Etot, 1, es2, ed2, mx2);
    edge_expsum_kernel<<<cdiv((long)Etot, 256), 256, 0, stream>>>(ei, E, Etot, 1, es2, ed2, mx2, dn2);
    edge_agg_kernel<<<cdiv((long)Etot * 32, 256), 256, 0, stream>>>(
        ei, E, Etot, 1, OUT, es2, ed2, mx2, dn2, g2, agg2);

    // 6) out = log_softmax(agg2 + b_last + skip)
    final_logsoftmax_kernel<<<cdiv((long)N * 32, 256), 256, 0, stream>>>(
        agg2, b_last, skip, out, N, OUT);
}
